// Duration_75265006895470
// MI455X (gfx1250) — matrix operand order repaired and checked by reading, not compiled, not run on hardware
//
#include <hip/hip_runtime.h>
#include <hip/hip_bf16.h>

#define BB 16
#define NN 512
#define EE 512
#define HH 256
#define GG 1024
#define TT 1024
#define PP 32
#define EP 544
#define M8 8192
#define LDA 544

typedef __attribute__((ext_vector_type(16))) __bf16 v16bf;
typedef __attribute__((ext_vector_type(8)))  float v8f;
typedef __attribute__((ext_vector_type(4)))  unsigned int v4u;

union V16u { v4u q[2]; v16bf v; };

__device__ __forceinline__ unsigned short f32_to_bf16(float f) {
  unsigned int u = __float_as_uint(f);
  u += 0x7FFFu + ((u >> 16) & 1u);
  return (unsigned short)(u >> 16);
}
__device__ __forceinline__ float sigf(float x) { return 1.0f / (1.0f + expf(-x)); }

// ---------------------------------------------------------------------------
// Generic batched WMMA GEMM: C[M,Nc] = A[M,K](bf16) * Bt[Nc,K](bf16)^T + b0+b1.
// 16x64 tile per wave (one A fragment -> 4 WMMAs) with a software-pipelined
// K loop: next k-slice's fragments are prefetched before the current slice's
// WMMAs so loads overlap the XDL pipe (partial s_wait_loadcnt, not full
// drains). Lane layouts per CDNA5 ISA 7.12.2.
// ---------------------------------------------------------------------------
__global__ __launch_bounds__(256)
void wmma_gemm_kernel(const unsigned short* __restrict__ A,
                      const unsigned short* __restrict__ Bt,
                      const float* __restrict__ bias0, const float* __restrict__ bias1,
                      float* __restrict__ C,
                      int M, int Nc, int Kpad, int lda, int ldb, int ldc,
                      long long sA, long long sB, long long sC)
{
  int wave = blockIdx.x * 8 + (threadIdx.x >> 5);
  int lane = threadIdx.x & 31;
  int ng = Nc >> 6;                        // number of 64-wide column groups
  int mt = wave / ng;
  int n64 = wave - mt * ng;
  if (mt * 16 >= M) return;                // uniform per wave -> EXEC all-1s
  long long b = blockIdx.y;
  A  += b * sA;  Bt += b * sB;  C += b * sC;
  const unsigned short* arow  = A  + (long long)(mt * 16 + (lane & 15)) * lda + ((lane >= 16) ? 8  : 0);
  const unsigned short* brow0 = Bt + (long long)(n64 * 64 + (lane & 15)) * ldb + ((lane >= 16) ? 8 : 0);
  v8f acc[4];
#pragma unroll
  for (int j = 0; j < 4; ++j) acc[j] = (v8f){0.f,0.f,0.f,0.f,0.f,0.f,0.f,0.f};

  // prologue: load slice k=0
  V16u a0, b0[4];
  a0.q[0] = *(const v4u*)(arow);
  a0.q[1] = *(const v4u*)(arow + 16);
#pragma unroll
  for (int j = 0; j < 4; ++j) {
    const unsigned short* brow = brow0 + (long long)j * 16 * ldb;
    b0[j].q[0] = *(const v4u*)(brow);
    b0[j].q[1] = *(const v4u*)(brow + 16);
  }
  // steady state: prefetch slice k0, then compute slice k0-32
  for (int k0 = 32; k0 < Kpad; k0 += 32) {
    V16u a1, b1[4];
    a1.q[0] = *(const v4u*)(arow + k0);
    a1.q[1] = *(const v4u*)(arow + k0 + 16);
#pragma unroll
    for (int j = 0; j < 4; ++j) {
      const unsigned short* brow = brow0 + (long long)j * 16 * ldb;
      b1[j].q[0] = *(const v4u*)(brow + k0);
      b1[j].q[1] = *(const v4u*)(brow + k0 + 16);
    }
#pragma unroll
    for (int j = 0; j < 4; ++j)
      acc[j] = __builtin_amdgcn_wmma_f32_16x16x32_bf16(false, a0.v, false, b0[j].v,
                                                       (short)0, acc[j], false, false);
    a0 = a1;
#pragma unroll
    for (int j = 0; j < 4; ++j) b0[j] = b1[j];
  }
  // epilogue: last slice
#pragma unroll
  for (int j = 0; j < 4; ++j)
    acc[j] = __builtin_amdgcn_wmma_f32_16x16x32_bf16(false, a0.v, false, b0[j].v,
                                                     (short)0, acc[j], false, false);

  int mbase = mt * 16 + ((lane >= 16) ? 8 : 0);
#pragma unroll
  for (int j = 0; j < 4; ++j) {
    int ncol = n64 * 64 + j * 16 + (lane & 15);
    float badd = 0.f;
    if (bias0) badd += bias0[ncol];
    if (bias1) badd += bias1[ncol];
#pragma unroll
    for (int v = 0; v < 8; ++v)
      C[(long long)(mbase + v) * ldc + ncol] = acc[j][v] + badd;
  }
}

// ---------------------------------------------------------------------------
// Persistent LSTM recurrence. grid.x: 0=forward, 1=backward (run concurrently).
// 512 threads = 16 waves; each wave owns one 16x64 gate tile group (64 tiles =
// 1024 gate columns; M=16 batches = exactly one WMMA tile row). Hidden state
// in LDS as bf16 [16][256]; same software-pipelined K loop (LDS A + global
// Whh B prefetched one slice ahead). Gate pre-activations staged f32
// [16][1024] in an overlapping, barrier-separated 64KB LDS region; cell
// state in registers.
// ---------------------------------------------------------------------------
__global__ __launch_bounds__(512)
void lstm_recurrent_kernel(const float* __restrict__ xg_f, const float* __restrict__ xg_b,
                           const unsigned short* __restrict__ whh_f,
                           const unsigned short* __restrict__ whh_b,
                           float* __restrict__ y, int Ntok)
{
  __shared__ alignas(16) char smem[65536];
  float* gates = (float*)smem;                    // [16][1024] f32 (64KB)
  unsigned short* hp = (unsigned short*)smem;     // [16][256] bf16 (8KB, overlapped)
  const bool rev = (blockIdx.x != 0);
  const float* xg = rev ? xg_b : xg_f;
  const unsigned short* whh = rev ? whh_b : whh_f;
  const int colbase = rev ? HH : 0;
  const int tid = threadIdx.x;
  const int lane = tid & 31;
  const int wv = tid >> 5;
  for (int i = tid; i < 16 * HH; i += 512) hp[i] = 0;
  float cst[8];
#pragma unroll
  for (int r = 0; r < 8; ++r) cst[r] = 0.f;
  __syncthreads();

  const unsigned short* arow  = hp + (lane & 15) * HH + ((lane >= 16) ? 8 : 0);
  const unsigned short* brow0 = whh + (long long)(wv * 64 + (lane & 15)) * HH + ((lane >= 16) ? 8 : 0);

  for (int s = 0; s < Ntok; ++s) {
    const int tok = rev ? (Ntok - 1 - s) : s;
    // Phase A: recurrent matmul, software-pipelined over 8 k-slices
    v8f acc[4];
#pragma unroll
    for (int tt = 0; tt < 4; ++tt) acc[tt] = (v8f){0.f,0.f,0.f,0.f,0.f,0.f,0.f,0.f};
    V16u a0, b0[4];
    a0.q[0] = *(const v4u*)(arow);
    a0.q[1] = *(const v4u*)(arow + 16);
#pragma unroll
    for (int tt = 0; tt < 4; ++tt) {
      const unsigned short* brow = brow0 + (long long)tt * 16 * HH;
      b0[tt].q[0] = *(const v4u*)(brow);
      b0[tt].q[1] = *(const v4u*)(brow + 16);
    }
    for (int k0 = 32; k0 < HH; k0 += 32) {
      V16u a1, b1[4];
      a1.q[0] = *(const v4u*)(arow + k0);
      a1.q[1] = *(const v4u*)(arow + k0 + 16);
#pragma unroll
      for (int tt = 0; tt < 4; ++tt) {
        const unsigned short* brow = brow0 + (long long)tt * 16 * HH;
        b1[tt].q[0] = *(const v4u*)(brow + k0);
        b1[tt].q[1] = *(const v4u*)(brow + k0 + 16);
      }
#pragma unroll
      for (int tt = 0; tt < 4; ++tt)
        acc[tt] = __builtin_amdgcn_wmma_f32_16x16x32_bf16(false, a0.v, false, b0[tt].v,
                                                          (short)0, acc[tt], false, false);
      a0 = a1;
#pragma unroll
      for (int tt = 0; tt < 4; ++tt) b0[tt] = b1[tt];
    }
#pragma unroll
    for (int tt = 0; tt < 4; ++tt)
      acc[tt] = __builtin_amdgcn_wmma_f32_16x16x32_bf16(false, a0.v, false, b0[tt].v,
                                                        (short)0, acc[tt], false, false);
    __syncthreads();   // all hp reads done before gates overwrite the region
    // Phase B: gates = recurrent + precomputed input gates
#pragma unroll
    for (int tt = 0; tt < 4; ++tt) {
      int ncol = wv * 64 + tt * 16 + (lane & 15);
#pragma unroll
      for (int v = 0; v < 8; ++v) {
        int m = v + ((lane >= 16) ? 8 : 0);
        gates[m * GG + ncol] = acc[tt][v] + xg[((long long)m * Ntok + tok) * GG + ncol];
      }
    }
    __syncthreads();
    // Phase C: LSTM cell update (torch gate order i,f,g,o)
    float hn[8];
#pragma unroll
    for (int r = 0; r < 8; ++r) {
      int idx = tid + r * 512;
      int bb = idx >> 8;
      int j  = idx & 255;
      float gi = gates[bb * GG + j];
      float gf = gates[bb * GG + HH + j];
      float gc = gates[bb * GG + 2 * HH + j];
      float go = gates[bb * GG + 3 * HH + j];
      float c  = sigf(gf) * cst[r] + sigf(gi) * tanhf(gc);
      cst[r] = c;
      float hh = sigf(go) * tanhf(c);
      hn[r] = hh;
      y[((long long)bb * Ntok + tok) * (2 * HH) + colbase + j] = hh;
    }
    __syncthreads();   // all gate reads done before hp overwrites region
    // Phase D: store new hidden state as bf16 for next step's A matrix
#pragma unroll
    for (int r = 0; r < 8; ++r) {
      int idx = tid + r * 512;
      int bb = idx >> 8;
      int j  = idx & 255;
      hp[bb * HH + j] = f32_to_bf16(hn[r]);
    }
    __syncthreads();
  }
}

// ---------------------------------------------------------------------------
// Small helper kernels
// ---------------------------------------------------------------------------
__global__ __launch_bounds__(512)
void prep_kernel(const float* __restrict__ d, float* __restrict__ dfr, float* __restrict__ center)
{
  __shared__ float s[NN];
  int b = blockIdx.x, n = threadIdx.x;
  float f = rintf(d[b * NN + n] * (22050.0f / 256.0f));   // round-to-even like jnp.round
  s[n] = f;
  __syncthreads();
  for (int off = 1; off < NN; off <<= 1) {
    float add = (n >= off) ? s[n - off] : 0.f;
    __syncthreads();
    s[n] += add;
    __syncthreads();
  }
  dfr[b * NN + n] = f;
  center[b * NN + n] = s[n] - 0.5f * f;
}

// pack f32 rows [M,Ksrc] (+optional appended scalar column) into bf16 [M,LDA]
__global__ void pack_A_kernel(const float* __restrict__ src, const float* __restrict__ append,
                              unsigned short* __restrict__ dst, int Ksrc, long long M)
{
  long long i = (long long)blockIdx.x * blockDim.x + threadIdx.x;
  if (i >= M * LDA) return;
  long long m = i / LDA;
  int k = (int)(i - m * LDA);
  float v = 0.f;
  if (k < Ksrc)                 v = src[m * Ksrc + k];
  else if (append && k == Ksrc) v = append[m];
  dst[i] = f32_to_bf16(v);
}

__global__ void pack_W_kernel(const float* __restrict__ W, unsigned short* __restrict__ dst,
                              int Ksrc, int Kdst, int G)
{
  long long i = (long long)blockIdx.x * blockDim.x + threadIdx.x;
  if (i >= (long long)G * Kdst) return;
  long long g = i / Kdst;
  int k = (int)(i - g * Kdst);
  dst[i] = f32_to_bf16((k < Ksrc) ? W[g * Ksrc + k] : 0.f);
}

// pred = y . W   (mode 0)  |  std = softplus(y . W) (mode 1); one wave per row
__global__ __launch_bounds__(256)
void proj_kernel(const float* __restrict__ y, const float* __restrict__ W,
                 float* __restrict__ dst, int mode)
{
  int wid = blockIdx.x * 8 + (threadIdx.x >> 5);
  int lane = threadIdx.x & 31;
  if (wid >= BB * NN) return;
  const float* row = y + (long long)wid * (2 * HH);
  float v = 0.f;
  for (int k = lane; k < 2 * HH; k += 32) v += row[k] * W[k];
#pragma unroll
  for (int off = 16; off > 0; off >>= 1) v += __shfl_xor(v, off, 32);
  if (lane == 0) {
    if (mode) v = (v > 20.f) ? v : log1pf(expf(v));
    dst[wid] = v;
  }
}

// normalized Gaussian attention weights, written as bf16 [B,T,N]
__global__ __launch_bounds__(256)
void upsample_w_kernel(const float* __restrict__ center, const float* __restrict__ stdv,
                       unsigned short* __restrict__ wub)
{
  __shared__ float red[256];
  int t = blockIdx.x, b = blockIdx.y, tid = threadIdx.x;
  float pos = (float)(t + 1);
  float ev[2]; float lsum = 0.f;
#pragma unroll
  for (int r = 0; r < 2; ++r) {
    int n = tid + r * 256;
    float df = pos - center[b * NN + n];
    float sd = stdv[b * NN + n];
    float ex = expf(-(df * df) / (sd * sd));
    ev[r] = ex; lsum += ex;
  }
  red[tid] = lsum;
  __syncthreads();
  for (int off = 128; off > 0; off >>= 1) {
    if (tid < off) red[tid] += red[tid + off];
    __syncthreads();
  }
  float inv = 1.0f / red[0];
#pragma unroll
  for (int r = 0; r < 2; ++r) {
    int n = tid + r * 256;
    wub[((long long)b * TT + t) * NN + n] = f32_to_bf16(ev[r] * inv);
  }
}

// hT[b][e][n] = bf16(h[b][n][e]) so the Ug GEMM's B loads are contiguous
__global__ void transpose_h_kernel(const float* __restrict__ h, unsigned short* __restrict__ hT)
{
  long long i = (long long)blockIdx.x * blockDim.x + threadIdx.x;
  if (i >= (long long)BB * EE * NN) return;
  int n = (int)(i % NN);
  long long be = i / NN;
  int e = (int)(be % EE);
  int b = (int)(be / EE);
  hT[i] = f32_to_bf16(h[((long long)b * NN + n) * EE + e]);
}

__global__ void pe_kernel(float* __restrict__ U)
{
  int i = blockIdx.x * blockDim.x + threadIdx.x;
  if (i >= BB * TT * PP) return;
  int p = i & (PP - 1);
  int t = (i >> 5) & (TT - 1);
  int b = i >> 15;
  float iv = (float)(2 * (p >> 1) + 1);             // 1,1,3,3,...,31,31
  float dv = powf(10000.0f, iv / 32.0f);
  float pos = (float)(t + 1);
  float val = (p & 1) ? cosf(pos / dv) : sinf(pos / dv);
  U[((long long)b * TT + t) * EP + EE + p] = val;
}

// ---------------------------------------------------------------------------
extern "C" void kernel_launch(void* const* d_in, const int* in_sizes, int n_in,
                              void* d_out, int out_size, void* d_ws, size_t ws_size,
                              hipStream_t stream)
{
  (void)in_sizes; (void)n_in; (void)out_size; (void)ws_size;
  const float* h    = (const float*)d_in[0];
  const float* dsec = (const float*)d_in[1];
  // d_in[2] = max_T (constant 1024)
  const float* LW[32];
  for (int i = 0; i < 32; ++i) LW[i] = (const float*)d_in[3 + i];  // dur L0, dur L1, rng L0, rng L1; each 8
  const float* W_dur   = (const float*)d_in[35];
  const float* W_range = (const float*)d_in[36];

  char* ws = (char*)d_ws;
  size_t off = 0;
  auto carve = [&](size_t bytes) -> void* {
    void* p = ws + off;
    off += (bytes + 255) & ~(size_t)255;
    return p;
  };
  float* dfr    = (float*)carve((size_t)BB * NN * 4);
  float* center = (float*)carve((size_t)BB * NN * 4);
  float* stdv   = (float*)carve((size_t)BB * NN * 4);
  unsigned short* xA    = (unsigned short*)carve((size_t)M8 * LDA * 2);
  unsigned short* wih_f = (unsigned short*)carve((size_t)GG * LDA * 2);
  unsigned short* wih_b = (unsigned short*)carve((size_t)GG * LDA * 2);
  unsigned short* whh_f = (unsigned short*)carve((size_t)GG * HH * 2);
  unsigned short* whh_b = (unsigned short*)carve((size_t)GG * HH * 2);
  float* xg_f = (float*)carve((size_t)M8 * GG * 4);
  float* xg_b = (float*)carve((size_t)M8 * GG * 4);
  float* yseq = (float*)carve((size_t)M8 * 2 * HH * 4);
  unsigned short* hT  = (unsigned short*)carve((size_t)BB * EE * NN * 2);
  unsigned short* wub = (unsigned short*)carve((size_t)BB * TT * NN * 2);

  float* U  = (float*)d_out;                       // [16,1024,544]
  float* pd = U + (size_t)BB * TT * EP;            // pred_duration [16,512]

  prep_kernel<<<BB, NN, 0, stream>>>(dsec, dfr, center);

  for (int br = 0; br < 2; ++br) {
    const float* const* LP = LW + br * 16;
    const float* append = (br == 0) ? nullptr : dfr;   // range branch appends d_frames
    for (int layer = 0; layer < 2; ++layer) {
      const float* const* Q = LP + layer * 8;          // Wif,Whf,bif,bhf,Wib,Whb,bib,bhb
      int KsrcW = (layer == 0 && append) ? 513 : 512;
      int Kdst  = (layer == 0 && append) ? 544 : 512;
      long long atot = (long long)M8 * LDA;
      pack_A_kernel<<<(int)((atot + 255) / 256), 256, 0, stream>>>(
          (layer == 0) ? h : yseq, (layer == 0) ? append : nullptr, xA, 512, (long long)M8);
      long long wtot = (long long)GG * Kdst;
      pack_W_kernel<<<(int)((wtot + 255) / 256), 256, 0, stream>>>(Q[0], wih_f, KsrcW, Kdst, GG);
      pack_W_kernel<<<(int)((wtot + 255) / 256), 256, 0, stream>>>(Q[4], wih_b, KsrcW, Kdst, GG);
      pack_W_kernel<<<(GG * HH + 255) / 256, 256, 0, stream>>>(Q[1], whh_f, HH, HH, GG);
      pack_W_kernel<<<(GG * HH + 255) / 256, 256, 0, stream>>>(Q[5], whh_b, HH, HH, GG);
      int tiles = (M8 / 16) * (GG / 64);               // 8192 waves (16x64 tiles)
      wmma_gemm_kernel<<<dim3(tiles / 8, 1), 256, 0, stream>>>(
          xA, wih_f, Q[2], Q[3], xg_f, M8, GG, Kdst, LDA, Kdst, GG, 0, 0, 0);
      wmma_gemm_kernel<<<dim3(tiles / 8, 1), 256, 0, stream>>>(
          xA, wih_b, Q[6], Q[7], xg_b, M8, GG, Kdst, LDA, Kdst, GG, 0, 0, 0);
      lstm_recurrent_kernel<<<2, 512, 0, stream>>>(xg_f, xg_b, whh_f, whh_b, yseq, NN);
    }
    if (br == 0)
      proj_kernel<<<M8 / 8, 256, 0, stream>>>(yseq, W_dur, pd, 0);
    else
      proj_kernel<<<M8 / 8, 256, 0, stream>>>(yseq, W_range, stdv, 1);
  }

  upsample_w_kernel<<<dim3(TT, BB), 256, 0, stream>>>(center, stdv, wub);
  transpose_h_kernel<<<(int)(((long long)BB * EE * NN + 255) / 256), 256, 0, stream>>>(h, hT);
  int tilesU = (TT / 16) * (EE / 64);                  // 512 waves per batch
  wmma_gemm_kernel<<<dim3(tilesU / 8, BB), 256, 0, stream>>>(
      wub, hT, nullptr, nullptr, U,
      TT, EE, NN, NN, NN, EP,
      (long long)TT * NN, (long long)EE * NN, (long long)TT * EP);
  pe_kernel<<<(BB * TT * PP) / 256, 256, 0, stream>>>(U);
}
